// OIM_MLP_22952305230318
// MI455X (gfx1250) — matrix operand order repaired and checked by reading, not compiled, not run on hardware
//
#include <hip/hip_runtime.h>

// ---------------------------------------------------------------------------
// OIM MLP phase descent on MI455X (gfx1250), wave32 + WMMA f16->f32.
// Sizes: B=128, n=[784,512,512,10], T=5, EPS=0.1
// GEMM kernel fuses output pairs sharing a B operand and uses 2x2 WMMA
// register blocking: 4 v_wmma per 8 b128 loads (32 B/WMMA vs 64 before).
// ---------------------------------------------------------------------------

#define BATCH   128
#define N0      784
#define N0P     800          // K padded to multiple of 32
#define N1      512
#define N2      512
#define N3      10
#define EPS     0.1f

typedef _Float16 v8h  __attribute__((ext_vector_type(8)));
typedef _Float16 v16h __attribute__((ext_vector_type(16)));
typedef float    v8f  __attribute__((ext_vector_type(8)));

// ---------------- prep kernels ----------------

__global__ void k_prep_x(const float* __restrict__ x, _Float16* __restrict__ Xp) {
    int i = blockIdx.x * blockDim.x + threadIdx.x;           // over 128*800
    if (i >= BATCH * N0P) return;
    int r = i / N0P, c = i % N0P;
    Xp[i] = (c < N0) ? (_Float16)x[r * N0 + c] : (_Float16)0.0f;
}

__global__ void k_prep_w0(const float* __restrict__ W0, _Float16* __restrict__ W0p) {
    int i = blockIdx.x * blockDim.x + threadIdx.x;           // over 512*800
    if (i >= N1 * N0P) return;
    int r = i / N0P, c = i % N0P;
    W0p[i] = (c < N0) ? (_Float16)W0[r * N0 + c] : (_Float16)0.0f;
}

__global__ void k_prep_w1(const float* __restrict__ W1,
                          _Float16* __restrict__ W1h, _Float16* __restrict__ W1Th) {
    int i = blockIdx.x * blockDim.x + threadIdx.x;           // over 512*512
    if (i >= N1 * N2) return;
    int r = i >> 9, c = i & 511;
    float v = W1[i];
    W1h[i] = (_Float16)v;                                    // W1h[j][i]  (for @W1^T)
    W1Th[c * N2 + r] = (_Float16)v;                          // W1Th[i][j] (for @W1)
}

__global__ void k_copy_phis(const float* __restrict__ p0, const float* __restrict__ p1,
                            const float* __restrict__ p2, float* __restrict__ out) {
    int i = blockIdx.x * blockDim.x + threadIdx.x;           // over 132352
    const int S0 = BATCH * N1, S1 = BATCH * (N1 + N2), S2 = BATCH * (N1 + N2) + BATCH * N3;
    if (i < S0)       out[i] = p0[i];
    else if (i < S1)  out[i] = p1[i - S0];
    else if (i < S2)  out[i] = p2[i - S1];
}

// ---------------- per-step trig ----------------

__global__ void k_trig(const float* __restrict__ phi0, const float* __restrict__ phi1,
                       const float* __restrict__ phi2,
                       _Float16* __restrict__ c0h, _Float16* __restrict__ s0h,
                       _Float16* __restrict__ c1h, _Float16* __restrict__ s1h,
                       float* __restrict__ c2t, float* __restrict__ s2t) {
    int i = blockIdx.x * blockDim.x + threadIdx.x;           // 65536 threads
    if (i >= BATCH * N1) return;
    float p = phi0[i];
    c0h[i] = (_Float16)__cosf(p); s0h[i] = (_Float16)__sinf(p);
    p = phi1[i];
    c1h[i] = (_Float16)__cosf(p); s1h[i] = (_Float16)__sinf(p);
    if (i < BATCH * N3) {
        p = phi2[i];
        c2t[i] = cosf(p); s2t[i] = sinf(p);
    }
}

// ---------------- WMMA fragment loaders ----------------
// A (row-major, lda halves): lane (r = lane&15, hi = lane>>4) takes
//   halves 0-7  = A[row0+r][k + 8*hi .. +7]
//   halves 8-15 = A[row0+r][k + 16 + 8*hi .. +7]
// B ("N-major K-contiguous", ldb halves): lane takes
//   halves 0-7 = B[col0+r][k + 8*hi .. +7], halves 8-15 = B[col0+r][k + 16 + 8*hi .. +7]

__device__ __forceinline__ v16h load_afrag(const _Float16* p) {
    v8h lo = *(const v8h*)p;
    v8h hi = *(const v8h*)(p + 16);
    return __builtin_shufflevector(lo, hi, 0,1,2,3,4,5,6,7,8,9,10,11,12,13,14,15);
}

__device__ __forceinline__ v16h load_bfrag(const _Float16* p) {
    v8h lo = *(const v8h*)p;
    v8h hi = *(const v8h*)(p + 16);
    return __builtin_shufflevector(lo, hi, 0,1,2,3,4,5,6,7,8,9,10,11,12,13,14,15);
}

#define WMMA_F16(A, B, C) \
    __builtin_amdgcn_wmma_f32_16x16x32_f16(false, (A), false, (B), (short)0, (C), false, false)

// ---------------- fused pair GEMM ----------------
// Computes C0 = A0*B and C1 = A1*B (shared B operand), f32 accumulate.
// Block = 256 threads = 8 waves. Each wave: one 16-row M tile x two 16-col
// N tiles for BOTH outputs => 4 accumulators, 4 WMMAs per 8 b128 loads.
// Grid: x = N/(8*32), y = M/16, z selects operand set.

struct GemmPairArgs {
    const _Float16* A0[2];
    const _Float16* A1[2];
    const _Float16* B[2];
    float*          C0[2];
    float*          C1[2];
};

__global__ __launch_bounds__(256) void k_gemm_pair(GemmPairArgs ga, int M, int N, int K,
                                                   int lda, int ldb) {
    const int z = blockIdx.z;
    const _Float16* __restrict__ A0 = ga.A0[z];
    const _Float16* __restrict__ A1 = ga.A1[z];
    const _Float16* __restrict__ B  = ga.B[z];
    float* __restrict__ C0 = ga.C0[z];
    float* __restrict__ C1 = ga.C1[z];

    const int wave = threadIdx.x >> 5;
    const int lane = threadIdx.x & 31;
    const int row0 = blockIdx.y * 16;
    const int col0 = (blockIdx.x * 8 + wave) * 32;           // two 16-wide N tiles
    if (row0 >= M || col0 >= N) return;                      // uniform per wave

    const int r  = lane & 15;
    const int hi = lane >> 4;

    const _Float16* a0p = A0 + (size_t)(row0 + r) * lda + (hi << 3);
    const _Float16* a1p = A1 + (size_t)(row0 + r) * lda + (hi << 3);
    const _Float16* b0p = B  + (size_t)(col0 + r) * ldb + (hi << 3);
    const _Float16* b1p = B  + (size_t)(col0 + 16 + r) * ldb + (hi << 3);

    v8f acc00 = {}, acc01 = {}, acc10 = {}, acc11 = {};
    for (int k = 0; k < K; k += 32) {
        v16h a0 = load_afrag(a0p + k);
        v16h a1 = load_afrag(a1p + k);
        v16h b0 = load_bfrag(b0p + k);
        v16h b1 = load_bfrag(b1p + k);
        acc00 = WMMA_F16(a0, b0, acc00);
        acc01 = WMMA_F16(a0, b1, acc01);
        acc10 = WMMA_F16(a1, b0, acc10);
        acc11 = WMMA_F16(a1, b1, acc11);
    }

    // C/D layout: VGPR v -> (M = v + 8*hi, N = lane&15)
    float* c0p = C0 + (size_t)(row0 + (hi << 3)) * N + col0 + r;
    float* c1p = C1 + (size_t)(row0 + (hi << 3)) * N + col0 + r;
#pragma unroll
    for (int v = 0; v < 8; ++v) {
        c0p[(size_t)v * N]      = acc00[v];
        c0p[(size_t)v * N + 16] = acc01[v];
        c1p[(size_t)v * N]      = acc10[v];
        c1p[(size_t)v * N + 16] = acc11[v];
    }
}

// ---------------- small W2^T dots: U = sinφ1 @ W2^T, V = cosφ1 @ W2^T  [128,10] -------

__global__ void k_gemm2(const _Float16* __restrict__ c1h, const _Float16* __restrict__ s1h,
                        const float* __restrict__ W2,
                        float* __restrict__ U, float* __restrict__ V) {
    int i = blockIdx.x * blockDim.x + threadIdx.x;           // 1280 threads
    if (i >= BATCH * N3) return;
    int b = i / N3, j = i % N3;
    const _Float16* sp = s1h + b * N2;
    const _Float16* cp = c1h + b * N2;
    const float*    wp = W2 + j * N2;
    float u = 0.f, v = 0.f;
    for (int t = 0; t < N2; ++t) {
        float w = wp[t];
        u += (float)sp[t] * w;
        v += (float)cp[t] * w;
    }
    U[i] = u; V[i] = v;
}

// ---------------- update kernels ----------------

__global__ void k_upd0(float* __restrict__ phi0, const float* __restrict__ a0,
                       const float* __restrict__ P, const float* __restrict__ Q,
                       const float* __restrict__ b0, const float* __restrict__ s0) {
    int i = blockIdx.x * blockDim.x + threadIdx.x;           // 65536
    if (i >= BATCH * N1) return;
    int col = i & 511;
    float p = phi0[i];
    float sv = sinf(p), cv = cosf(p);
    float g = sv * (a0[i] + P[i] + b0[col]) - cv * Q[i] + 2.f * sv * cv * s0[col];
    phi0[i] = p - EPS * g;
}

__global__ void k_upd1(float* __restrict__ phi1, const float* __restrict__ R,
                       const float* __restrict__ S, const float* __restrict__ c2t,
                       const float* __restrict__ s2t, const float* __restrict__ W2,
                       const float* __restrict__ b1, const float* __restrict__ s1) {
    int i = blockIdx.x * blockDim.x + threadIdx.x;           // 65536
    if (i >= BATCH * N2) return;
    int b = i >> 9, col = i & 511;
    const float* c2p = c2t + b * N3;
    const float* s2p = s2t + b * N3;
    float cw = 0.f, sw = 0.f;
#pragma unroll
    for (int j = 0; j < N3; ++j) {
        float w = W2[j * N2 + col];
        cw += c2p[j] * w;                                    // (cosφ2 @ W2)
        sw += s2p[j] * w;                                    // (sinφ2 @ W2)
    }
    float p = phi1[i];
    float sv = sinf(p), cv = cosf(p);
    float g = sv * (S[i] + cw + b1[col]) - cv * (R[i] + sw) + 2.f * sv * cv * s1[col];
    phi1[i] = p - EPS * g;
}

__global__ void k_upd2(float* __restrict__ phi2, const float* __restrict__ U,
                       const float* __restrict__ V, const float* __restrict__ b2,
                       const float* __restrict__ s2in) {
    int i = blockIdx.x * blockDim.x + threadIdx.x;           // 1280
    if (i >= BATCH * N3) return;
    int j = i % N3;
    float p = phi2[i];
    float sv = sinf(p), cv = cosf(p);
    float g = -cv * U[i] + sv * (V[i] + b2[j]) + 2.f * sv * cv * s2in[j];
    phi2[i] = p - EPS * g;
}

// ---------------- launch ----------------

extern "C" void kernel_launch(void* const* d_in, const int* in_sizes, int n_in,
                              void* d_out, int out_size, void* d_ws, size_t ws_size,
                              hipStream_t stream) {
    const float* x    = (const float*)d_in[0];
    const float* W0   = (const float*)d_in[1];
    const float* W1   = (const float*)d_in[2];
    const float* W2   = (const float*)d_in[3];
    const float* b0   = (const float*)d_in[4];
    const float* b1   = (const float*)d_in[5];
    const float* b2   = (const float*)d_in[6];
    const float* s0   = (const float*)d_in[7];
    const float* s1   = (const float*)d_in[8];
    const float* s2   = (const float*)d_in[9];
    const float* p0in = (const float*)d_in[10];
    const float* p1in = (const float*)d_in[11];
    const float* p2in = (const float*)d_in[12];
    // d_in[13]=y (unused, beta=0), d_in[14]=T (=5 in reference setup)

    float* out  = (float*)d_out;
    float* phi0 = out;                       // [128,512]
    float* phi1 = out + BATCH * N1;          // [128,512]
    float* phi2 = out + BATCH * (N1 + N2);   // [128,10]

    // ---- workspace carve-up (halves first, then f32; all 16B aligned) ----
    _Float16* wsH  = (_Float16*)d_ws;
    _Float16* Xp   = wsH;                       // 128*800
    _Float16* W0p  = Xp   + BATCH * N0P;        // 512*800
    _Float16* W1h  = W0p  + N1 * N0P;           // 512*512
    _Float16* W1Th = W1h  + N1 * N2;            // 512*512
    _Float16* c0h  = W1Th + N1 * N2;            // 128*512 each below
    _Float16* s0h  = c0h  + BATCH * N1;
    _Float16* c1h  = s0h  + BATCH * N1;
    _Float16* s1h  = c1h  + BATCH * N1;
    float* wsF  = (float*)(s1h + BATCH * N1);
    float* a0   = wsF;                          // 128*512 each below
    float* P    = a0   + BATCH * N1;
    float* Q    = P    + BATCH * N1;
    float* R    = Q    + BATCH * N1;
    float* S    = R    + BATCH * N1;
    float* sink = S    + BATCH * N1;            // duplicate-output sink (128*512)
    float* c2t  = sink + BATCH * N1;            // 128*10 each below
    float* s2t  = c2t  + BATCH * N3;
    float* U    = s2t  + BATCH * N3;
    float* V    = U    + BATCH * N3;

    const int TB = 256;
    // ---- one-time prep ----
    k_prep_x <<<(BATCH * N0P + TB - 1) / TB, TB, 0, stream>>>(x, Xp);
    k_prep_w0<<<(N1 * N0P   + TB - 1) / TB, TB, 0, stream>>>(W0, W0p);
    k_prep_w1<<<(N1 * N2    + TB - 1) / TB, TB, 0, stream>>>(W1, W1h, W1Th);
    k_copy_phis<<<(BATCH * (N1 + N2 + N3) + TB - 1) / TB, TB, 0, stream>>>(p0in, p1in, p2in, out);

    // a0 = x @ W0^T  (M=128, N=512, K=800 padded); second output -> sink
    GemmPairArgs gA;
    gA.A0[0] = Xp;  gA.A1[0] = Xp;  gA.B[0] = W0p; gA.C0[0] = a0; gA.C1[0] = sink;
    gA.A0[1] = Xp;  gA.A1[1] = Xp;  gA.B[1] = W0p; gA.C0[1] = a0; gA.C1[1] = sink;
    k_gemm_pair<<<dim3(N1 / 256, BATCH / 16, 1), TB, 0, stream>>>(gA, BATCH, N1, N0P, N0P, N0P);

    // z=0: {P,Q} = {cosφ1, sinφ1} @ W1   (B = W1Th layout)
    // z=1: {R,S} = {sinφ0, cosφ0} @ W1^T (B = W1h  layout)
    GemmPairArgs gS;
    gS.A0[0] = c1h; gS.A1[0] = s1h; gS.B[0] = W1Th; gS.C0[0] = P; gS.C1[0] = Q;
    gS.A0[1] = s0h; gS.A1[1] = c0h; gS.B[1] = W1h;  gS.C0[1] = R; gS.C1[1] = S;

    for (int t = 0; t < 5; ++t) {
        k_trig<<<(BATCH * N1 + TB - 1) / TB, TB, 0, stream>>>(
            phi0, phi1, phi2, c0h, s0h, c1h, s1h, c2t, s2t);
        k_gemm_pair<<<dim3(N1 / 256, BATCH / 16, 2), TB, 0, stream>>>(gS, BATCH, N1, N2, N2, N2);
        k_gemm2<<<(BATCH * N3 + TB - 1) / TB, TB, 0, stream>>>(c1h, s1h, W2, U, V);
        k_upd0<<<(BATCH * N1 + TB - 1) / TB, TB, 0, stream>>>(phi0, a0, P, Q, b0, s0);
        k_upd1<<<(BATCH * N2 + TB - 1) / TB, TB, 0, stream>>>(phi1, R, S, c2t, s2t, W2, b1, s1);
        k_upd2<<<(BATCH * N3 + TB - 1) / TB, TB, 0, stream>>>(phi2, U, V, b2, s2);
    }
}
